// GumbelQuantizer_9328668967809
// MI455X (gfx1250) — matrix operand order repaired and checked by reading, not compiled, not run on hardware
//
#include <hip/hip_runtime.h>
#include <hip/hip_bf16.h>

typedef __attribute__((ext_vector_type(8)))  __bf16 v8bf;
typedef __attribute__((ext_vector_type(16))) __bf16 v16bf;
typedef __attribute__((ext_vector_type(8)))  float  v8f;

#define NROWS 16384
#define DDIM  512
#define KCODES 1024
#define A_STRIDE 520   // 512 + 8 bf16 pad to break LDS bank alignment

// ---------------------------------------------------------------- utilities
__device__ __forceinline__ v16bf ld_frag(const __bf16* p0, const __bf16* p1) {
    v8bf a = *(const v8bf*)p0;
    v8bf b = *(const v8bf*)p1;
    return __builtin_shufflevector(a, b, 0,1,2,3,4,5,6,7,8,9,10,11,12,13,14,15);
}

// ---------------------------------------------------------------- kernel 1
__global__ void zero_counts_kernel(unsigned int* __restrict__ counts) {
    counts[blockIdx.x * 256 + threadIdx.x] = 0u;
}

// ---------------------------------------------------------------- kernel 2
// codebook (1024 x 512 f32) -> bf16 hi/lo copies + exact f32 ||c||^2
__global__ __launch_bounds__(256) void prep_codebook_kernel(
    const float* __restrict__ cb,
    __bf16* __restrict__ cb_hi, __bf16* __restrict__ cb_lo,
    float* __restrict__ c2) {
    const int wave = threadIdx.x >> 5;
    const int lane = threadIdx.x & 31;
    const int code = blockIdx.x * 8 + wave;           // grid 128 -> 1024 codes
    const float* row = cb + (size_t)code * DDIM;
    float ss = 0.f;
    #pragma unroll
    for (int j = 0; j < 16; ++j) {
        const int k = lane + j * 32;
        const float v = row[k];
        const __bf16 h = (__bf16)v;
        cb_hi[(size_t)code * DDIM + k] = h;
        cb_lo[(size_t)code * DDIM + k] = (__bf16)(v - (float)h);
        ss += v * v;
    }
    #pragma unroll
    for (int m = 16; m >= 1; m >>= 1) ss += __shfl_xor(ss, m, 32);
    if (lane == 0) c2[code] = ss;
}

// ---------------------------------------------------------------- kernel 3
// Fused GEMM (bf16x3 WMMA) + softmax + argmax + q-gather + histogram.
// Block = 256 threads (8 waves); block tile = 16 rows x 1024 cols;
// wave tile = 16 rows x 128 cols = 8 WMMA accumulators.
__global__ __launch_bounds__(256) void vq_main_kernel(
    const float* __restrict__ x, const float* __restrict__ cb,
    const __bf16* __restrict__ cb_hi, const __bf16* __restrict__ cb_lo,
    const float* __restrict__ c2, unsigned int* __restrict__ counts,
    float* __restrict__ q_out, float* __restrict__ probs_out) {

    __shared__ __bf16 aHs[16 * A_STRIDE];
    __shared__ __bf16 aLs[16 * A_STRIDE];
    __shared__ float  red_val[8 * 16];
    __shared__ int    red_idx[8 * 16];
    __shared__ float  rowmax[16];
    __shared__ int    rowarg[16];
    __shared__ float  rowinv[16];

    const int tid  = threadIdx.x;
    const int row0 = blockIdx.x * 16;

    // ---- Phase 1: stage 16 rows of x into LDS as bf16 hi/lo
    #pragma unroll
    for (int j = 0; j < 32; ++j) {
        const int e = tid + j * 256;          // 0..8191
        const int r = e >> 9, k = e & 511;
        const float v = x[(size_t)(row0 + r) * DDIM + k];
        const __bf16 h = (__bf16)v;
        aHs[r * A_STRIDE + k] = h;
        aLs[r * A_STRIDE + k] = (__bf16)(v - (float)h);
    }
    __syncthreads();

    const int wave = tid >> 5, lane = tid & 31;
    const int l15 = lane & 15, half = lane >> 4;
    const int colbase = wave * 128;

    const v8f zero8 = {0.f,0.f,0.f,0.f,0.f,0.f,0.f,0.f};
    v8f acc[8];
    #pragma unroll
    for (int t = 0; t < 8; ++t) acc[t] = zero8;

    // ---- Phase 2: GEMM over D=512 in 16 chunks of K=32
    for (int kc = 0; kc < 16; ++kc) {
        // A fragment (16x32 bf16): lane -> row l15, elems [kb,kb+8) ++ [kb+16,kb+24)
        const __bf16* ah = &aHs[l15 * A_STRIDE + kc * 32 + half * 8];
        const __bf16* al = &aLs[l15 * A_STRIDE + kc * 32 + half * 8];
        const v16bf aH = ld_frag(ah, ah + 16);
        const v16bf aL = ld_frag(al, al + 16);
        #pragma unroll
        for (int t = 0; t < 8; ++t) {
            // B fragment (32x16): lane -> col, elems [kb,kb+8) ++ [kb+16,kb+24)
            const size_t boff = (size_t)(colbase + t * 16 + l15) * DDIM
                              + kc * 32 + half * 8;
            const v16bf bH = ld_frag(cb_hi + boff, cb_hi + boff + 16);
            const v16bf bL = ld_frag(cb_lo + boff, cb_lo + boff + 16);
            acc[t] = __builtin_amdgcn_wmma_f32_16x16x32_bf16(
                false, aH, false, bH, (short)0, acc[t], false, false);
            acc[t] = __builtin_amdgcn_wmma_f32_16x16x32_bf16(
                false, aH, false, bL, (short)0, acc[t], false, false);
            acc[t] = __builtin_amdgcn_wmma_f32_16x16x32_bf16(
                false, aL, false, bH, (short)0, acc[t], false, false);
        }
    }

    // ---- Phase 3: scores s = 2*dot - ||c||^2 ; per-row max/argmax
    float c2v[8];
    #pragma unroll
    for (int t = 0; t < 8; ++t) c2v[t] = c2[colbase + t * 16 + l15];

    #pragma unroll
    for (int i = 0; i < 8; ++i) {            // slot i: row i (lanes 0-15) / i+8 (16-31)
        float best = -3.4e38f; int bidx = 0;
        #pragma unroll
        for (int t = 0; t < 8; ++t) {
            const float s = 2.0f * acc[t][i] - c2v[t];
            acc[t][i] = s;
            if (s > best) { best = s; bidx = colbase + t * 16 + l15; }
        }
        #pragma unroll
        for (int m = 1; m <= 8; m <<= 1) {   // reduce within each 16-lane half
            const float ov = __shfl_xor(best, m, 32);
            const int   oi = __shfl_xor(bidx, m, 32);
            if (ov > best || (ov == best && oi < bidx)) { best = ov; bidx = oi; }
        }
        if (l15 == 0) {
            const int r = i + half * 8;
            red_val[wave * 16 + r] = best;
            red_idx[wave * 16 + r] = bidx;
        }
    }
    __syncthreads();
    if (tid < 16) {                           // cross-wave argmax, first-index ties
        float best = red_val[tid]; int bidx = red_idx[tid];
        for (int w = 1; w < 8; ++w) {
            const float v = red_val[w * 16 + tid];
            const int  ix = red_idx[w * 16 + tid];
            if (v > best || (v == best && ix < bidx)) { best = v; bidx = ix; }
        }
        rowmax[tid] = best; rowarg[tid] = bidx;
    }
    __syncthreads();

    // ---- Phase 4: exp + row sums
    #pragma unroll
    for (int i = 0; i < 8; ++i) {
        const int r = i + half * 8;
        const float m = rowmax[r];
        float ps = 0.f;
        #pragma unroll
        for (int t = 0; t < 8; ++t) {
            const float e = __expf(acc[t][i] - m);
            acc[t][i] = e;
            ps += e;
        }
        #pragma unroll
        for (int mk = 1; mk <= 8; mk <<= 1) ps += __shfl_xor(ps, mk, 32);
        if (l15 == 0) red_val[wave * 16 + r] = ps;
    }
    __syncthreads();
    if (tid < 16) {
        float s = 0.f;
        for (int w = 0; w < 8; ++w) s += red_val[w * 16 + tid];
        rowinv[tid] = 1.0f / s;
    }
    __syncthreads();

    // ---- Phase 5: write normalized probs straight from accumulators
    #pragma unroll
    for (int i = 0; i < 8; ++i) {
        const int r = i + half * 8;
        const float inv = rowinv[r];
        #pragma unroll
        for (int t = 0; t < 8; ++t) {
            const int col = colbase + t * 16 + l15;
            probs_out[(size_t)(row0 + r) * KCODES + col] = acc[t][i] * inv;
        }
    }

    // ---- Phase 6: histogram + q = codebook[argmax]
    if (tid < 16) atomicAdd(&counts[rowarg[tid]], 1u);
    {
        const int r = tid >> 4;               // 16 threads per row
        const int t16 = tid & 15;
        const int idx = rowarg[r];
        const float4* src = (const float4*)(cb + (size_t)idx * DDIM);
        float4* dst = (float4*)(q_out + (size_t)(row0 + r) * DDIM);
        #pragma unroll
        for (int j = 0; j < 8; ++j) dst[t16 + j * 16] = src[t16 + j * 16];
    }
}

// ---------------------------------------------------------------- kernel 4
__global__ void perp_kernel(const unsigned int* __restrict__ counts,
                            float* __restrict__ out) {
    __shared__ float sb[256];
    const int tid = threadIdx.x;
    float s = 0.f;
    #pragma unroll
    for (int j = 0; j < 4; ++j) {
        const float p = (float)counts[tid + j * 256] * (1.0f / (float)NROWS);
        s += p * logf(p + 1e-12f);
    }
    sb[tid] = s;
    __syncthreads();
    for (int m = 128; m >= 1; m >>= 1) {
        if (tid < m) sb[tid] += sb[tid + m];
        __syncthreads();
    }
    if (tid == 0) out[0] = __expf(-sb[0]);
}

// ---------------------------------------------------------------- launcher
extern "C" void kernel_launch(void* const* d_in, const int* in_sizes, int n_in,
                              void* d_out, int out_size, void* d_ws, size_t ws_size,
                              hipStream_t stream) {
    (void)in_sizes; (void)n_in; (void)out_size; (void)ws_size;
    const float* x  = (const float*)d_in[0];   // (8,2048,512) f32
    const float* cb = (const float*)d_in[1];   // (1024,512)   f32

    char* ws = (char*)d_ws;
    __bf16* cb_hi = (__bf16*)ws;                               // 1 MB
    __bf16* cb_lo = (__bf16*)(ws + (size_t)1048576);           // 1 MB
    float*  c2    = (float*) (ws + (size_t)2097152);           // 4 KB
    unsigned int* counts = (unsigned int*)(ws + (size_t)2097152 + 4096); // 4 KB

    float* q_out     = (float*)d_out;                          // 16384*512
    float* probs_out = q_out + (size_t)NROWS * DDIM;           // 16384*1024
    float* perp_out  = probs_out + (size_t)NROWS * KCODES;     // 1

    zero_counts_kernel<<<4, 256, 0, stream>>>(counts);
    prep_codebook_kernel<<<128, 256, 0, stream>>>(cb, cb_hi, cb_lo, c2);
    vq_main_kernel<<<NROWS / 16, 256, 0, stream>>>(x, cb, cb_hi, cb_lo, c2,
                                                   counts, q_out, probs_out);
    perp_kernel<<<1, 256, 0, stream>>>(counts, perp_out);
}
